// GatedRecurrentCell_79577154060722
// MI455X (gfx1250) — matrix operand order repaired and checked by reading, compiled, not run on hardware
//
#include <hip/hip_runtime.h>

// ---------------------------------------------------------------------------
// Problem constants (from reference): B=8, S=2048, D=512, I=2048
// ---------------------------------------------------------------------------
#define B_DIM 8
#define S_DIM 2048
#define D_DIM 512
#define I_DIM 2048
#define M_TOTAL (B_DIM * S_DIM)          // 16384 rows of the flattened GEMM

typedef __attribute__((ext_vector_type(16))) __bf16 v16bf;
typedef __attribute__((ext_vector_type(8)))  __bf16 v8bf;
typedef __attribute__((ext_vector_type(8)))  float  v8f;
typedef __attribute__((ext_vector_type(4)))  int    v4i;

typedef __attribute__((address_space(3))) unsigned char lds_uchar;
// async-copy builtin pointer types (from the compiler diagnostic):
//   param0: int4* in AS(1) (global), param1: int4* in AS(3) (LDS)
typedef __attribute__((address_space(1))) v4i* gptr_v4i;
typedef __attribute__((address_space(3))) v4i* lptr_v4i;

// LDS staging geometry: per matrix tile = 64 cols x 32 K x 2B (= 4 KB data)
#define LDS_COL_STRIDE 80                        // 64B data + 16B pad (16B aligned)
#define LDS_MAT_BYTES  (64 * LDS_COL_STRIDE)     // 5120 B
#define LDS_BUF_BYTES  (2 * LDS_MAT_BYTES)       // 10240 B (Wa tile, then Wi tile)

#if __has_builtin(__builtin_amdgcn_global_load_async_to_lds_b128)
#define USE_ASYNC_LDS 1
#else
#define USE_ASYNC_LDS 0
#endif

__device__ __forceinline__ void wait_asynccnt0() {
#if __has_builtin(__builtin_amdgcn_s_wait_asynccnt)
    __builtin_amdgcn_s_wait_asynccnt(0);
#else
    asm volatile("s_wait_asynccnt 0x0" ::: "memory");
#endif
}

// ---------------------------------------------------------------------------
// fp32 -> bf16 (round to nearest even), stored as ushort
// ---------------------------------------------------------------------------
__device__ __forceinline__ unsigned short f2bf(float f) {
    union { float f; unsigned int u; } v;
    v.f = f;
    unsigned int u = v.u;
    u += 0x7FFFu + ((u >> 16) & 1u);
    return (unsigned short)(u >> 16);
}

__global__ __launch_bounds__(256)
void cvt_bf16_kernel(const float* __restrict__ src, unsigned short* __restrict__ dst, int n4) {
    int t = blockIdx.x * blockDim.x + threadIdx.x;
    if (t < n4) {
        const float4 f = *(const float4*)(src + (size_t)t * 4);
        ushort4 o;
        o.x = f2bf(f.x); o.y = f2bf(f.y); o.z = f2bf(f.z); o.w = f2bf(f.w);
        *(ushort4*)(dst + (size_t)t * 4) = o;
    }
}

// ---------------------------------------------------------------------------
// Fused dual-GEMM + gate epilogue; weight tiles staged to LDS per workgroup
// (double-buffered) via gfx1250 async global->LDS copies when available.
//   pa = x @ Wa^T + ba ; pi = x @ Wi^T + bi   (bf16 WMMA, f32 accumulate)
//   a  = sigmoid(gate) * 3^{-sigmoid(pa)}     -> ws
//   c  = sqrt(1-a^2) * sigmoid(pi) * pi       -> d_out
// Grid: x = M/128 supertiles (8 waves x 16 rows), y = N/64 strips.
// ---------------------------------------------------------------------------
__global__ __launch_bounds__(256)
void gemm_gate_kernel(const __bf16* __restrict__ xh,
                      const __bf16* __restrict__ wah,
                      const __bf16* __restrict__ wih,
                      const float*  __restrict__ ba,
                      const float*  __restrict__ bi,
                      const float*  __restrict__ gate,
                      float* __restrict__ a_arr,
                      float* __restrict__ c_arr) {
    __shared__ __align__(16) unsigned char lds[2 * LDS_BUF_BYTES];   // 20 KB

    const int tid   = threadIdx.x;
    const int lane  = tid & 31;
    const int wave  = tid >> 5;
    const int mtile = blockIdx.x * 128 + wave * 16;   // row base of 16-row tile
    const int nstrip = blockIdx.y * 64;               // col base of 64-col strip

    const int ln15 = lane & 15;
    const int hi   = lane >> 4;      // 0 for lanes 0-15, 1 for 16-31
    const int akb  = hi * 16;        // A-fragment K base (contiguous K order, matches B)

    // ---- cooperative B staging: each thread owns one (matrix, col, half) ---
    // 32B of one weight column per thread per K-step (two 16B pieces).
    const int smat  = tid >> 7;          // 0 = Wa, 1 = Wi
    const int scol  = (tid & 127) >> 1;  // 0..63
    const int shalf = tid & 1;           // K half: 0 -> K 0..15, 1 -> K 16..31
    __bf16* gsrc = (__bf16*)(smat ? wih : wah)
                 + (size_t)(nstrip + scol) * D_DIM + shalf * 16;
    const int sdst_off = smat * LDS_MAT_BYTES + scol * LDS_COL_STRIDE + shalf * 32;
    lds_uchar* sdst = (lds_uchar*)(lds + sdst_off);

    // ---- A fragment source: row (mtile+ln15), two 16B chunks per K step ----
    const __bf16* arow = xh + (size_t)(mtile + ln15) * D_DIM + akb;

    // ---- per-wave LDS fragment read offsets --------------------------------
    const int frag_off = ln15 * LDS_COL_STRIDE + hi * 32;

    v8f acca[4], acci[4];
#pragma unroll
    for (int j = 0; j < 4; ++j) { acca[j] = (v8f)(0.0f); acci[j] = (v8f)(0.0f); }

#if USE_ASYNC_LDS
    // ---- prologue: async-stage K-step 0 ------------------------------------
    __builtin_amdgcn_global_load_async_to_lds_b128(
        (gptr_v4i)(gsrc),     (lptr_v4i)(sdst),      0, 0);
    __builtin_amdgcn_global_load_async_to_lds_b128(
        (gptr_v4i)(gsrc + 8), (lptr_v4i)(sdst + 16), 0, 0);
    wait_asynccnt0();
#else
    v16bf greg = *(const v16bf*)gsrc;
    {
        v8bf glo = __builtin_shufflevector(greg, greg, 0, 1, 2, 3, 4, 5, 6, 7);
        v8bf ghi = __builtin_shufflevector(greg, greg, 8, 9, 10, 11, 12, 13, 14, 15);
        *(v8bf*)(lds + sdst_off)      = glo;
        *(v8bf*)(lds + sdst_off + 16) = ghi;
    }
#endif
    __syncthreads();
    v8bf alo = *(const v8bf*)(arow);
    v8bf ahi = *(const v8bf*)(arow + 8);

    // ---- main K loop: 16 steps of K=32, double-buffered --------------------
    for (int t = 0; t < 16; ++t) {
        const int kt = t * 32;
        unsigned char* cur = lds + (t & 1) * LDS_BUF_BYTES;

        v16bf afrag = __builtin_shufflevector(alo, ahi,
                        0, 1, 2, 3, 4, 5, 6, 7, 8, 9, 10, 11, 12, 13, 14, 15);

        // prefetch next K-step while we compute this one
        if (t < 15) {
#if USE_ASYNC_LDS
            lds_uchar* ndst = sdst + ((t + 1) & 1) * LDS_BUF_BYTES;
            __builtin_amdgcn_global_load_async_to_lds_b128(
                (gptr_v4i)(gsrc + kt + 32), (lptr_v4i)(ndst),      0, 0);
            __builtin_amdgcn_global_load_async_to_lds_b128(
                (gptr_v4i)(gsrc + kt + 40), (lptr_v4i)(ndst + 16), 0, 0);
#else
            greg = *(const v16bf*)(gsrc + kt + 32);
#endif
            alo = *(const v8bf*)(arow + kt + 32);
            ahi = *(const v8bf*)(arow + kt + 40);
        }

        // read the 8 B fragments of this K-step from LDS
        v16bf bfa[4], bfi[4];
#pragma unroll
        for (int j = 0; j < 4; ++j) {
            const unsigned char* pa8 = cur + j * (16 * LDS_COL_STRIDE) + frag_off;
            v8bf l0 = *(const v8bf*)(pa8);
            v8bf h0 = *(const v8bf*)(pa8 + 16);
            bfa[j] = __builtin_shufflevector(l0, h0,
                        0, 1, 2, 3, 4, 5, 6, 7, 8, 9, 10, 11, 12, 13, 14, 15);
            const unsigned char* pi8 = pa8 + LDS_MAT_BYTES;
            v8bf l1 = *(const v8bf*)(pi8);
            v8bf h1 = *(const v8bf*)(pi8 + 16);
            bfi[j] = __builtin_shufflevector(l1, h1,
                        0, 1, 2, 3, 4, 5, 6, 7, 8, 9, 10, 11, 12, 13, 14, 15);
        }

#pragma unroll
        for (int j = 0; j < 4; ++j)
            acca[j] = __builtin_amdgcn_wmma_f32_16x16x32_bf16(
                          false, afrag, false, bfa[j], (short)0, acca[j], false, false);
#pragma unroll
        for (int j = 0; j < 4; ++j)
            acci[j] = __builtin_amdgcn_wmma_f32_16x16x32_bf16(
                          false, afrag, false, bfi[j], (short)0, acci[j], false, false);

        // make next buffer visible to the whole workgroup
        if (t < 15) {
#if USE_ASYNC_LDS
            wait_asynccnt0();
#else
            unsigned char* nxt = lds + ((t + 1) & 1) * LDS_BUF_BYTES;
            v8bf glo = __builtin_shufflevector(greg, greg, 0, 1, 2, 3, 4, 5, 6, 7);
            v8bf ghi = __builtin_shufflevector(greg, greg, 8, 9, 10, 11, 12, 13, 14, 15);
            *(v8bf*)(nxt + sdst_off)      = glo;
            *(v8bf*)(nxt + sdst_off + 16) = ghi;
#endif
            __syncthreads();
        }
    }

    // ---- epilogue: C/D layout = VGPR v -> row M = v + 8*hi, lane&15 -> col N
    const float LN3 = 1.0986122886681098f;
    const int rowbase = mtile + hi * 8;
#pragma unroll
    for (int j = 0; j < 4; ++j) {
        const int col = nstrip + 16 * j + ln15;
        const float alpha = 1.0f / (1.0f + __expf(-gate[col]));
        const float bav = ba[col];
        const float biv = bi[col];
#pragma unroll
        for (int v = 0; v < 8; ++v) {
            const float pav = acca[j][v] + bav;
            const float piv = acci[j][v] + biv;
            const float rg = 1.0f / (1.0f + __expf(-pav));   // reset gate
            const float ig = 1.0f / (1.0f + __expf(-piv));   // input gate
            const float aa = alpha * __expf(-rg * LN3);      // alpha / 3^rg
            const float dr = __fsqrt_rn(1.0f - aa * aa) * (ig * piv);
            const size_t idx = (size_t)(rowbase + v) * I_DIM + col;
            a_arr[idx] = aa;
            c_arr[idx] = dr;
        }
    }
}

// ---------------------------------------------------------------------------
// Sequential scan over S, in place on d_out. One lane per (b, i) chain:
// 16384 chains, consecutive lanes -> consecutive i -> fully coalesced.
// ---------------------------------------------------------------------------
__global__ __launch_bounds__(256)
void scan_kernel(const float* __restrict__ a_arr, float* __restrict__ out) {
    const int t = blockIdx.x * blockDim.x + threadIdx.x;  // 0..16383
    const int b = t >> 11;                                 // / I_DIM
    const int i = t & (I_DIM - 1);
    size_t base = (size_t)b * S_DIM * I_DIM + i;
    float h = 0.0f;
    for (int s = 0; s < S_DIM; s += 4) {
        const float a0 = a_arr[base];
        const float c0 = out[base];
        const float a1 = a_arr[base + I_DIM];
        const float c1 = out[base + I_DIM];
        const float a2 = a_arr[base + 2 * I_DIM];
        const float c2 = out[base + 2 * I_DIM];
        const float a3 = a_arr[base + 3 * I_DIM];
        const float c3 = out[base + 3 * I_DIM];
        h = fmaf(a0, h, c0); out[base] = h;
        h = fmaf(a1, h, c1); out[base + I_DIM] = h;
        h = fmaf(a2, h, c2); out[base + 2 * I_DIM] = h;
        h = fmaf(a3, h, c3); out[base + 3 * I_DIM] = h;
        base += 4 * (size_t)I_DIM;
    }
}

// ---------------------------------------------------------------------------
// Launcher
// ---------------------------------------------------------------------------
extern "C" void kernel_launch(void* const* d_in, const int* in_sizes, int n_in,
                              void* d_out, int out_size, void* d_ws, size_t ws_size,
                              hipStream_t stream) {
    const float* x    = (const float*)d_in[0];   // [B,S,D]
    const float* Wa   = (const float*)d_in[1];   // [I,D]
    const float* ba   = (const float*)d_in[2];   // [I]
    const float* Wi   = (const float*)d_in[3];   // [I,D]
    const float* bi   = (const float*)d_in[4];   // [I]
    const float* gate = (const float*)d_in[5];   // [I]
    float* out = (float*)d_out;                  // [B,S,I] — holds c, then h

    // Workspace layout
    const size_t A_ELEMS = (size_t)M_TOTAL * I_DIM;          // 33.5M f32 = 134 MB
    const size_t X_ELEMS = (size_t)M_TOTAL * D_DIM;          // 8.39M bf16
    const size_t W_ELEMS = (size_t)I_DIM * D_DIM;            // 1.05M bf16
    char* ws = (char*)d_ws;
    float*          a_arr = (float*)ws;
    unsigned short* xh    = (unsigned short*)(ws + A_ELEMS * sizeof(float));
    unsigned short* wah   = xh + X_ELEMS;
    unsigned short* wih   = wah + W_ELEMS;

    // 1) fp32 -> bf16 conversions
    {
        int n4 = (int)(X_ELEMS / 4);
        cvt_bf16_kernel<<<(n4 + 255) / 256, 256, 0, stream>>>(x, xh, n4);
        n4 = (int)(W_ELEMS / 4);
        cvt_bf16_kernel<<<(n4 + 255) / 256, 256, 0, stream>>>(Wa, wah, n4);
        cvt_bf16_kernel<<<(n4 + 255) / 256, 256, 0, stream>>>(Wi, wih, n4);
    }

    // 2) fused dual-GEMM + gate epilogue (WMMA bf16, async-LDS staged weights)
    {
        dim3 grid(M_TOTAL / 128, I_DIM / 64);  // 128 x 32
        gemm_gate_kernel<<<grid, 256, 0, stream>>>(
            (const __bf16*)xh, (const __bf16*)wah, (const __bf16*)wih,
            ba, bi, gate, a_arr, out);
    }

    // 3) sequential recurrence, in place on d_out (B*I = 16384 lanes)
    scan_kernel<<<64, 256, 0, stream>>>(a_arr, out);
}
